// GlobalAttention_60129542717
// MI455X (gfx1250) — matrix operand order repaired and checked by reading, not compiled, not run on hardware
//
#include <hip/hip_runtime.h>
#include <hip/hip_bf16.h>
#include <math.h>

// ---------------------------------------------------------------------------
// GlobalAttention for MI455X (gfx1250, wave32, WMMA 16x16x32 f16)
//
// Dominant cost: rw = att @ windows = 275 GFLOP -> fused flash-style single
// pass (reference exp has no max-subtraction, so numerator+denominator
// accumulate together). All GEMMs via v_wmma_f32_16x16x32_f16 with f32
// accumulation; B operands fed from pre-transposed f16 buffers so each lane's
// B fragment is one contiguous 32B load (ISA B-layout: lane=column, K pairs
// packed per VGPR).
//
// Attention tile: M=32 queries x 2048 cols per workgroup (512 thr, 16 waves
// x 128 cols). Each winT B-fragment feeds two WMMAs (two query sub-tiles),
// halving L2 read traffic vs M=16 (16 GB -> 8 GB) while keeping 128
// accumulator VGPRs per wave.
// ---------------------------------------------------------------------------

typedef __attribute__((ext_vector_type(16))) _Float16 v16h;
typedef __attribute__((ext_vector_type(8)))  _Float16 v8h;
typedef __attribute__((ext_vector_type(8)))  float    v8f;

#define EPSF  1e-8f
#define DIMS  2048      // ws*ws*dim
#define NQ    16384     // total windows (B*N)
#define NKEY  4096      // windows per batch
#define NB    4
#define EMB   32        // embedding padded 30 -> 32

__device__ __forceinline__ v8f wmma16x16x32(v16h a, v16h b, v8f c) {
  return __builtin_amdgcn_wmma_f32_16x16x32_f16(false, a, false, b,
                                                (short)0, c, false, false);
}

// --------------------------- source stats ----------------------------------
// Per-batch sum / sumsq over 8.39M elements; 512 blocks/batch write partials
// (deterministic fixed-order finalize later; no FP atomics).
__global__ __launch_bounds__(256) void k_src_stats(const float* __restrict__ win,
                                                   float* __restrict__ srcPart) {
  const int b = blockIdx.y;
  const size_t base = (size_t)b * ((size_t)NKEY * DIMS) + (size_t)blockIdx.x * 16384;
  float s = 0.f, q = 0.f;
  for (int i = threadIdx.x; i < 16384; i += 256) {
    float v = win[base + i];
    s += v; q += v * v;
  }
  __shared__ float rs[256], rq[256];
  rs[threadIdx.x] = s; rq[threadIdx.x] = q;
  __syncthreads();
  for (int off = 128; off > 0; off >>= 1) {
    if (threadIdx.x < off) {
      rs[threadIdx.x] += rs[threadIdx.x + off];
      rq[threadIdx.x] += rq[threadIdx.x + off];
    }
    __syncthreads();
  }
  if (threadIdx.x == 0) {
    int blk = b * 512 + blockIdx.x;
    srcPart[blk * 2 + 0] = rs[0];
    srcPart[blk * 2 + 1] = rq[0];
  }
}

// ------------------- transpose + f32->f16 convert --------------------------
// in: f32 [R][C] row-major (per blockIdx.z slice), out: f16 [C][R].
// Gives WMMA B-operands contiguous per-lane 32B loads.
__global__ __launch_bounds__(256) void k_transpose_cvt(const float* __restrict__ in,
                                                       _Float16* __restrict__ out,
                                                       int R, int C) {
  __shared__ float tile[32][33];
  const size_t zo = (size_t)blockIdx.z * (size_t)R * (size_t)C;
  const float* inz = in + zo;
  _Float16* outz = out + zo;
  const int c0 = blockIdx.x * 32, r0 = blockIdx.y * 32;
  const int tx = threadIdx.x, ty = threadIdx.y;
#pragma unroll
  for (int i = 0; i < 4; ++i)
    tile[ty + i * 8][tx] = inz[(size_t)(r0 + ty + i * 8) * C + (c0 + tx)];
  __syncthreads();
#pragma unroll
  for (int i = 0; i < 4; ++i) {
    int oc = ty + i * 8;
    outz[(size_t)(c0 + oc) * R + (r0 + tx)] = (_Float16)tile[tx][oc];
  }
}

// ------------------------------ embedding ----------------------------------
// 16 rows per workgroup. GEMM1 (2048->256) via WMMA: 8 waves x 32 hidden cols,
// K-loop of 64 chunks; A built inline from f32 windows (per-lane float4 loads
// in contiguous k order 16h+s), B from pre-transposed W1T. GEMM2
// (256->30) is 0.25 GFLOP -> scalar over LDS. Then l2-norm, f16 store (padded
// to 32), and per-block cur-stats partials.
__global__ __launch_bounds__(256) void k_embed(const float* __restrict__ win,
                                               const _Float16* __restrict__ w1t,
                                               const float* __restrict__ b1,
                                               const float* __restrict__ W2,
                                               const float* __restrict__ b2,
                                               _Float16* __restrict__ xe,
                                               float* __restrict__ curPart) {
  __shared__ float h_lds[16][264];
  __shared__ float x2_lds[16][32];
  __shared__ float redS[16], redQ[16];
  const int q0  = blockIdx.x * 16;
  const int L   = threadIdx.x & 31;
  const int w   = threadIdx.x >> 5;
  const int l16 = L & 15;
  const int hi  = L >> 4;
  const int kb  = hi ? 16 : 0;           // contiguous k order: lanes>=16 own K 16-31
  const int col0 = w * 32 + l16;
  const int col1 = col0 + 16;
  const float* arow = win + (size_t)(q0 + l16) * DIMS;

  v8f c0 = {}; v8f c1 = {};
  for (int k0 = 0; k0 < DIMS; k0 += 32) {
    v16h a;
    const float4* p0 = (const float4*)(arow + k0 + kb);
    const float4* p1 = (const float4*)(arow + k0 + kb + 8);
    float4 f0 = p0[0], f1 = p0[1], g0 = p1[0], g1 = p1[1];
    a[0]  = (_Float16)f0.x; a[1]  = (_Float16)f0.y; a[2]  = (_Float16)f0.z; a[3]  = (_Float16)f0.w;
    a[4]  = (_Float16)f1.x; a[5]  = (_Float16)f1.y; a[6]  = (_Float16)f1.z; a[7]  = (_Float16)f1.w;
    a[8]  = (_Float16)g0.x; a[9]  = (_Float16)g0.y; a[10] = (_Float16)g0.z; a[11] = (_Float16)g0.w;
    a[12] = (_Float16)g1.x; a[13] = (_Float16)g1.y; a[14] = (_Float16)g1.z; a[15] = (_Float16)g1.w;
    const int klo = k0 + (hi ? 16 : 0);  // B, same contiguous order: lanes>=16 own K 16-31
    v16h bb0 = *(const v16h*)(w1t + (size_t)col0 * DIMS + klo);
    v16h bb1 = *(const v16h*)(w1t + (size_t)col1 * DIMS + klo);
    c0 = wmma16x16x32(a, bb0, c0);
    c1 = wmma16x16x32(a, bb1, c1);
  }
  // bias + relu, scatter C layout (lane=col, VGPR r = row) into LDS
#pragma unroll
  for (int r = 0; r < 8; ++r) {
    int m = r + hi * 8;
    float v0 = c0[r] + b1[col0];
    float v1 = c1[r] + b1[col1];
    h_lds[m][col0] = v0 > 0.f ? v0 : 0.f;
    h_lds[m][col1] = v1 > 0.f ? v1 : 0.f;
  }
  __syncthreads();
  // GEMM2: 16 rows x 30 cols, dots of length 256
  {
    int row = threadIdx.x & 15;
    int j0  = threadIdx.x >> 4;
    for (int jj = j0; jj < 30; jj += 16) {
      float s = b2[jj];
      for (int c = 0; c < 256; ++c) s += h_lds[row][c] * W2[c * 30 + jj];
      x2_lds[row][jj] = s;
    }
  }
  __syncthreads();
  if (threadIdx.x < 16) {
    int r = threadIdx.x;
    float ss = 0.f;
    for (int j = 0; j < 30; ++j) ss += x2_lds[r][j] * x2_lds[r][j];
    float inv = 1.0f / (sqrtf(ss) + EPSF);      // eps AFTER sqrt (matches ref)
    _Float16* xr = xe + (size_t)(q0 + r) * EMB;
    float s1 = 0.f, s2 = 0.f;
    for (int j = 0; j < 30; ++j) {
      float v = x2_lds[r][j] * inv;
      xr[j] = (_Float16)v;
      s1 += v; s2 += v * v;
    }
    xr[30] = (_Float16)0.f; xr[31] = (_Float16)0.f;  // K padding -> exact
    redS[r] = s1; redQ[r] = s2;
  }
  __syncthreads();
  if (threadIdx.x == 0) {
    float s = 0.f, q = 0.f;
    for (int i = 0; i < 16; ++i) { s += redS[i]; q += redQ[i]; }
    curPart[blockIdx.x * 2 + 0] = s;
    curPart[blockIdx.x * 2 + 1] = q;
  }
}

// ------------------------------ finalize -----------------------------------
// Fixed-order reduction of partials; folds AdaIN into out = rw*g + h0.
__global__ void k_finalize(const float* __restrict__ srcPart,
                           const float* __restrict__ curPart,
                           float* __restrict__ stats,
                           const float* __restrict__ alpha,
                           const float* __restrict__ beta) {
  if (threadIdx.x != 0) return;
  for (int b = 0; b < NB; ++b) {
    float s = 0.f, q = 0.f;
    for (int i = 0; i < 512; ++i) {
      s += srcPart[(b * 512 + i) * 2 + 0];
      q += srcPart[(b * 512 + i) * 2 + 1];
    }
    const float n1 = (float)((size_t)NKEY * DIMS);
    float sm = s / n1;
    float sv = q / n1 - sm * sm;
    float sstd = sqrtf(fmaxf(sv, 0.f) + EPSF);
    float cs = 0.f, cq = 0.f;
    for (int i = 0; i < 256; ++i) {
      int blk = b * 256 + i;
      cs += curPart[blk * 2 + 0];
      cq += curPart[blk * 2 + 1];
    }
    const float n2 = (float)(NKEY * 30);
    float cm = cs / n2;
    float cv = cq / n2 - cm * cm;
    float cstd = sqrtf(fmaxf(cv, 0.f) + EPSF);
    float g = sstd / cstd;
    stats[16 + b] = g;
    stats[20 + b] = sm - cm * g;
  }
  stats[24] = 30.f * alpha[0];   // ATT_A * alpha
  stats[25] = 20.f * beta[0];    // ATT_B * beta
}

// ------------------------- fused attention ---------------------------------
// Workgroup: 32 queries x 2048 output cols; 16 waves x 128 cols. Per wave:
// 2 query sub-tiles x 8 col tiles = 128 f32 accumulator VGPRs. Single pass
// over 4096 keys (chunk=32):
//   logits (4 WMMA: 2 query sub-tiles x 2 key sub-tiles) -> exp -> f16 att
//   staged through per-wave LDS to convert C-layout -> A-layout (padded
//   stride 40 => aligned b128 readback, same-wave RAW closed with
//   s_wait_dscnt) -> 8 col tiles, each winT B-fragment feeding 2 WMMAs.
// Denominator partials accumulate during readback; combined via LDS at end.
__global__ __launch_bounds__(512) void k_attn(const _Float16* __restrict__ xe,
                                              const _Float16* __restrict__ winT,
                                              const float* __restrict__ stats,
                                              float* __restrict__ out) {
  __shared__ __align__(16) _Float16 att_lds[16][32][40];
  __shared__ float den_lds[16][2][32];
  const int q0  = blockIdx.x * 32;
  const int b   = q0 >> 12;
  const int w   = threadIdx.x >> 5;      // wave 0..15
  const int L   = threadIdx.x & 31;
  const int l16 = L & 15;
  const int hi  = L >> 4;
  const int kb  = hi ? 16 : 0;           // contiguous k order: lanes>=16 own K 16-31
  const int dof = hi ? 16 : 0;           // B, same contiguous order: lanes>=16 own K 16-31
  const int cwave = w * 128;
  const float cA = stats[24], cB = stats[25];
  const float gG = stats[16 + b], hH = stats[20 + b];
  const _Float16* xkeys = xe + (size_t)b * NKEY * EMB;
  const _Float16* wT    = winT + (size_t)b * (size_t)DIMS * NKEY;

  // query A fragments (two 16-row sub-tiles), fixed for whole kernel
  v16h aq0, aq1;
  {
    const _Float16* xr0 = xe + (size_t)(q0 + l16) * EMB;
    const _Float16* xr1 = xe + (size_t)(q0 + 16 + l16) * EMB;
    *((v8h*)&aq0 + 0) = *(const v8h*)(xr0 + kb);
    *((v8h*)&aq0 + 1) = *(const v8h*)(xr0 + kb + 8);
    *((v8h*)&aq1 + 0) = *(const v8h*)(xr1 + kb);
    *((v8h*)&aq1 + 1) = *(const v8h*)(xr1 + kb + 8);
  }

  v8f acc0[8], acc1[8];
#pragma unroll
  for (int t = 0; t < 8; ++t) { acc0[t] = (v8f){}; acc1[t] = (v8f){}; }
  float ds0 = 0.f, ds1 = 0.f;

  for (int k0 = 0; k0 < NKEY; k0 += 32) {
    // logits: q[32x32] . keys^T -> four 16x16 tiles
    v16h bk0 = *(const v16h*)(xkeys + (size_t)(k0 + l16) * EMB + dof);
    v16h bk1 = *(const v16h*)(xkeys + (size_t)(k0 + 16 + l16) * EMB + dof);
    v8f z = {};
    v8f l00 = wmma16x16x32(aq0, bk0, z);
    v8f l01 = wmma16x16x32(aq0, bk1, z);
    v8f l10 = wmma16x16x32(aq1, bk0, z);
    v8f l11 = wmma16x16x32(aq1, bk1, z);
    // exp + stage att (f16) in per-wave LDS: C layout -> row-major [32q][32k]
#pragma unroll
    for (int r = 0; r < 8; ++r) {
      int m = r + hi * 8;
      att_lds[w][m][l16]           = (_Float16)__expf(cA * l00[r] + cB);
      att_lds[w][m][16 + l16]      = (_Float16)__expf(cA * l01[r] + cB);
      att_lds[w][16 + m][l16]      = (_Float16)__expf(cA * l10[r] + cB);
      att_lds[w][16 + m][16 + l16] = (_Float16)__expf(cA * l11[r] + cB);
    }
    asm volatile("s_wait_dscnt 0" ::: "memory");  // same-wave LDS RAW
    // rebuild as A operands + denominator partials
    v16h aa0, aa1;
    {
      const _Float16* rp0 = &att_lds[w][l16][0];
      const _Float16* rp1 = &att_lds[w][16 + l16][0];
      v8h t0 = *(const v8h*)(rp0 + kb);
      v8h t1 = *(const v8h*)(rp0 + kb + 8);
      v8h u0 = *(const v8h*)(rp1 + kb);
      v8h u1 = *(const v8h*)(rp1 + kb + 8);
      *((v8h*)&aa0 + 0) = t0;
      *((v8h*)&aa0 + 1) = t1;
      *((v8h*)&aa1 + 0) = u0;
      *((v8h*)&aa1 + 1) = u1;
#pragma unroll
      for (int j = 0; j < 8; ++j) {
        ds0 += (float)t0[j] + (float)t1[j];
        ds1 += (float)u0[j] + (float)u1[j];
      }
    }
    // numerator: att @ winT columns; each B fragment feeds 2 WMMAs
#pragma unroll
    for (int t = 0; t < 8; ++t) {
      const _Float16* wp = wT + (size_t)(cwave + t * 16 + l16) * NKEY + k0 + dof;
      v16h bw = *(const v16h*)wp;
      acc0[t] = wmma16x16x32(aa0, bw, acc0[t]);
      acc1[t] = wmma16x16x32(aa1, bw, acc1[t]);
    }
  }
  den_lds[w][0][L] = ds0;
  den_lds[w][1][L] = ds1;
  __syncthreads();
  // normalize + AdaIN fold + store
#pragma unroll
  for (int t = 0; t < 8; ++t) {
    const int col = cwave + t * 16 + l16;
#pragma unroll
    for (int r = 0; r < 8; ++r) {
      int m = r + hi * 8;
      float den0 = den_lds[w][0][m] + den_lds[w][0][m + 16] + EPSF;
      float den1 = den_lds[w][1][m] + den_lds[w][1][m + 16] + EPSF;
      out[(size_t)(q0 + m) * DIMS + col]      = acc0[t][r] / den0 * gG + hH;
      out[(size_t)(q0 + 16 + m) * DIMS + col] = acc1[t][r] / den1 * gG + hH;
    }
  }
}

// ---------------------------------------------------------------------------
extern "C" void kernel_launch(void* const* d_in, const int* in_sizes, int n_in,
                              void* d_out, int out_size, void* d_ws, size_t ws_size,
                              hipStream_t stream) {
  (void)in_sizes; (void)n_in; (void)out_size; (void)ws_size;
  const float* windows = (const float*)d_in[0];
  const float* W1      = (const float*)d_in[1];
  const float* b1    = (const float*)d_in[2];
  const float* W2    = (const float*)d_in[3];
  const float* b2    = (const float*)d_in[4];
  const float* alpha = (const float*)d_in[5];
  const float* beta  = (const float*)d_in[6];
  float* out = (float*)d_out;

  // workspace layout (all WMMA bases 32B aligned)
  char* ws = (char*)d_ws;
  float*    stats   = (float*)ws;                         //   256 B  (g,h0,cA,cB)
  float*    srcPart = (float*)(ws + 256);                 // 16384 B  (4x512x2)
  float*    curPart = (float*)(ws + 256 + 16384);         //  8192 B  (1024x2)
  _Float16* xe      = (_Float16*)(ws + 24832);            // 16384*32*2 = 1 MiB
  _Float16* w1t     = (_Float16*)(ws + 24832 + 1048576);  // 256*2048*2 = 1 MiB
  _Float16* winT    = (_Float16*)(ws + 24832 + 2 * 1048576); // 4*2048*4096*2 = 64 MiB

  // 1) per-batch source stats partials
  k_src_stats<<<dim3(512, NB), 256, 0, stream>>>(windows, srcPart);
  // 2) windows f32 [4096][2048] -> winT f16 [2048][4096] per batch
  k_transpose_cvt<<<dim3(2048 / 32, 4096 / 32, NB), dim3(32, 8), 0, stream>>>(
      windows, winT, 4096, 2048);
  // 3) W1 f32 [2048][256] -> W1T f16 [256][2048]
  k_transpose_cvt<<<dim3(256 / 32, 2048 / 32, 1), dim3(32, 8), 0, stream>>>(
      W1, w1t, 2048, 256);
  // 4) embedding MLP + l2norm -> xe, cur-stats partials
  k_embed<<<NQ / 16, 256, 0, stream>>>(windows, w1t, b1, W2, b2, xe, curPart);
  // 5) fold stats into per-batch scale/bias + attention coefficients
  k_finalize<<<1, 1, 0, stream>>>(srcPart, curPart, stats, alpha, beta);
  // 6) fused single-pass attention (exp/normalize/AdaIN folded; M=32)
  k_attn<<<NQ / 32, 512, 0, stream>>>(xe, winT, stats, out);
}
